// XCA_label_1_37082747634063
// MI455X (gfx1250) — matrix operand order repaired and checked by reading, not compiled, not run on hardware
//
#include <hip/hip_runtime.h>
#include <hip/hip_bf16.h>

typedef __attribute__((ext_vector_type(16))) _Float16 v16h;
typedef __attribute__((ext_vector_type(8)))  float    v8f;
typedef __attribute__((ext_vector_type(4)))  unsigned int u32x4;
typedef __attribute__((ext_vector_type(4)))  int       i32x4;
typedef __attribute__((ext_vector_type(8)))  int       i32x8;

#define B_  4
#define N_  2048
#define C_  1024
#define H_  16
#define D_  64
#define BN_ (B_*N_)

__device__ __forceinline__ v8f wmma32(v16h a, v16h b, v8f c) {
  // v_wmma_f32_16x16x32_f16 : D = A(16x32) * B(32x16) + C, fp32 accumulate
  return __builtin_amdgcn_wmma_f32_16x16x32_f16(false, a, false, b, (short)0, c,
                                                false, false);
}

// ---- TDM: async 4KB contiguous tile load into LDS (TENSORcnt tracked) -------
// D# per CDNA5 ISA: group0 {count=1, lds_addr,
// global_addr, type=2}; group1 {data_size=8B, tensor_dim0=512, tensor_dim1=1,
// tile_dim0=512, tile_dim1=1, tensor_dim0_stride=512}; groups 2/3 unused (2D).
__device__ __forceinline__ void tdm_load_4kb(unsigned lds_off, const void* gptr) {
  const unsigned long long ga = (unsigned long long)(uintptr_t)gptr;
  u32x4 g0;
  g0[0] = 1u;                                              // count=1, user desc
  g0[1] = lds_off;                                         // lds_addr (bytes)
  g0[2] = (unsigned)(ga & 0xFFFFFFFFu);                    // global_addr[31:0]
  g0[3] = (unsigned)((ga >> 32) & 0x01FFFFFFu) | (2u << 30); // [56:32] | type=2
  i32x8 g1;
  g1[0] = (3 << 16);                 // workgroup_mask=0, data_size=3 (8 bytes)
  g1[1] = (int)(512u << 16);         // tensor_dim0[15:0]=512 at bits[63:48]
  g1[2] = (int)(1u << 16);           // tensor_dim0[31:16]=0, tensor_dim1[15:0]=1
  g1[3] = (int)(512u << 16);         // tensor_dim1 hi=0, tile_dim0=512 (=4KB)
  g1[4] = 1;                         // tile_dim1=1, tile_dim2=0
  g1[5] = 512;                       // tensor_dim0_stride[31:0]
  g1[6] = 0;                         // stride hi / dim1_stride lo
  g1[7] = 0;
  const i32x4 z4 = {0, 0, 0, 0};
#if __clang_major__ >= 23
  const i32x8 z8 = {0, 0, 0, 0, 0, 0, 0, 0};
  __builtin_amdgcn_tensor_load_to_lds(g0, g1, z4, z4, z8, 0);
#else
  __builtin_amdgcn_tensor_load_to_lds(g0, g1, z4, z4, 0);
#endif
}

// ---- fragment loaders -------------------------------------------------------
// A fragment (16x32 f16). Lanes 0-15 hold rows 0-15 with
// K = 0..15 (elems 0..15); lanes 16-31 hold K = 16..31.
__device__ __forceinline__ v16h load_a_f32(const float* __restrict__ src, int ld,
                                           int rowbase, int k0) {
  const int lane = threadIdx.x & 31;
  const float* p = src + (size_t)(rowbase + (lane & 15)) * ld + k0 + ((lane >> 4) << 4);
  v16h a;
#pragma unroll
  for (int i = 0; i < 8; ++i) { a[i] = (_Float16)p[i]; a[8 + i] = (_Float16)p[8 + i]; }
  return a;
}
__device__ __forceinline__ v16h load_a_f16(const _Float16* __restrict__ src, int ld,
                                           int rowbase, int k0) {
  const int lane = threadIdx.x & 31;
  const _Float16* p = src + (size_t)(rowbase + (lane & 15)) * ld + k0 + ((lane >> 4) << 4);
  v16h a;
#pragma unroll
  for (int i = 0; i < 8; ++i) { a[i] = p[i]; a[8 + i] = p[8 + i]; }
  return a;
}
// B fragment (32x16 f16) where B[k][n] is stored as M[n][k] (each lane reads a
// contiguous 16-element run of one "column" row). Lanes 0-15: K = k0..k0+15,
// lanes 16-31: K = k0+16..k0+31; col = colbase + (lane&15).
__device__ __forceinline__ v16h load_b_colmaj_f32(const float* __restrict__ M, int ld,
                                                  int colbase, int k0) {
  const int lane = threadIdx.x & 31;
  const float* p = M + (size_t)(colbase + (lane & 15)) * ld + k0 + ((lane >> 4) << 4);
  v16h b;
#pragma unroll
  for (int i = 0; i < 16; ++i) b[i] = (_Float16)p[i];
  return b;
}
__device__ __forceinline__ v16h load_b_colmaj_f16(const _Float16* __restrict__ M, int ld,
                                                  int colbase, int k0) {
  const int lane = threadIdx.x & 31;
  const _Float16* p = M + (size_t)(colbase + (lane & 15)) * ld + k0 + ((lane >> 4) << 4);
  v16h b;
#pragma unroll
  for (int i = 0; i < 16; ++i) b[i] = p[i];
  return b;
}

// ---- K1: fused QKV projection + L2 norm + temperature, f16 out --------------
// grid (BN/64, H, 3), block 128 (4 waves, 16 rows each). Wave computes a
// 16 x 64 slab (= one head's full D) so the row norm is an in-wave reduction.
__global__ void __launch_bounds__(128)
proj_qkv_kernel(const float* __restrict__ x, const float* __restrict__ y,
                const float* __restrict__ q_w, const float* __restrict__ kv_w,
                const float* __restrict__ temperature,
                _Float16* __restrict__ Qn, _Float16* __restrict__ Kn,
                _Float16* __restrict__ Vh) {
  const int wave = threadIdx.x >> 5;
  const int lane = threadIdx.x & 31;
  const int lh = lane >> 4, lr = lane & 15;
  const int rowbase = blockIdx.x * 64 + wave * 16;   // row in flattened (B*N)
  const int h = blockIdx.y;
  const int which = blockIdx.z;                      // 0=Q 1=K 2=V

  const float* src = (which == 0) ? x : y;
  const float* W   = (which == 0) ? q_w : kv_w;
  const int wrow   = h * D_ + ((which == 2) ? C_ : 0);  // kv: o = (s*H + h)*D + d

  v8f acc[4] = {};
  for (int k0 = 0; k0 < C_; k0 += 32) {
    v16h A = load_a_f32(src, C_, rowbase, k0);
#pragma unroll
    for (int t = 0; t < 4; ++t)
      acc[t] = wmma32(A, load_b_colmaj_f32(W, C_, wrow + 16 * t, k0), acc[t]);
  }

  float scale[8];
  if (which < 2) {
#pragma unroll
    for (int i = 0; i < 8; ++i) {
      float sq = acc[0][i] * acc[0][i] + acc[1][i] * acc[1][i] +
                 acc[2][i] * acc[2][i] + acc[3][i] * acc[3][i];
#pragma unroll
      for (int m = 1; m <= 8; m <<= 1) sq += __shfl_xor(sq, m, 32);
      scale[i] = 1.0f / fmaxf(sqrtf(sq), 1e-12f);
    }
    if (which == 0) {
      const float tp = temperature[h];
#pragma unroll
      for (int i = 0; i < 8; ++i) scale[i] *= tp;   // fold temperature into Q
    }
  } else {
#pragma unroll
    for (int i = 0; i < 8; ++i) scale[i] = 1.0f;
  }

  _Float16* dst = (which == 0) ? Qn : (which == 1) ? Kn : Vh;
  const int b  = rowbase >> 11;          // / N_
  const int nb = rowbase & (N_ - 1);
  _Float16* dp = dst + (size_t)(b * H_ + h) * N_ * D_;
#pragma unroll
  for (int t = 0; t < 4; ++t)
#pragma unroll
    for (int i = 0; i < 8; ++i) {
      const int n = nb + i + lh * 8;
      dp[(size_t)n * D_ + 16 * t + lr] = (_Float16)(acc[t][i] * scale[i]);
    }
}

// ---- K2: flash-attention core, TDM double-buffered K, scrambled Z out -------
// grid (N/128, H, B), block 256 (8 waves; each wave owns 16 query rows).
__global__ void __launch_bounds__(256)
xca_attn_kernel(const _Float16* __restrict__ Qn, const _Float16* __restrict__ Kn,
                const _Float16* __restrict__ Vh, _Float16* __restrict__ Z) {
  __shared__ _Float16 Ks2[2][32 * 64];   // K chunk double buffer (TDM target)
  __shared__ _Float16 Vt[64 * 32];       // V chunk transposed (d, key)
  __shared__ _Float16 Ps[8][16 * 32];    // per-wave P staging

  const int wave = threadIdx.x >> 5;
  const int lane = threadIdx.x & 31;
  const int lh = lane >> 4, lr = lane & 15;
  const int h = blockIdx.y, b = blockIdx.z;
  const int qbase = blockIdx.x * 128 + wave * 16;
  const int NJ = N_ / 32;

  const size_t hdbase = (size_t)(b * H_ + h) * N_ * D_;
  const _Float16* qp = Qn + hdbase;
  const _Float16* kp = Kn + hdbase;
  const _Float16* vp = Vh + hdbase;

  const v16h Aq0 = load_a_f16(qp, D_, qbase, 0);
  const v16h Aq1 = load_a_f16(qp, D_, qbase, 32);

  float mrow[8], lrow[8];
  v8f O[4] = {};
#pragma unroll
  for (int i = 0; i < 8; ++i) { mrow[i] = -3.0e38f; lrow[i] = 0.0f; }

  // flat-address low 32 bits == LDS byte offset (ISA aperture rule)
  const unsigned ks_lds0 = (unsigned)(uintptr_t)&Ks2[0][0];
  const unsigned ks_lds1 = (unsigned)(uintptr_t)&Ks2[1][0];

  if (wave == 0) tdm_load_4kb(ks_lds0, kp);          // prime chunk 0

  for (int j = 0; j < NJ; ++j) {
    // stage V chunk transposed (manual; TDM cannot transpose)
    {
      const int kk = threadIdx.x >> 3;
      const int db = (threadIdx.x & 7) * 8;
      const _Float16* vs = vp + (size_t)(j * 32 + kk) * D_ + db;
#pragma unroll
      for (int i = 0; i < 8; ++i) Vt[(db + i) * 32 + kk] = vs[i];
    }
    if (j + 1 < NJ)  // global_prefetch_b8 for the next V chunk
      __builtin_prefetch(vp + (size_t)(j + 1) * 32 * D_ + (size_t)threadIdx.x * 8, 0, 1);

    if (wave == 0) {
      if (j + 1 < NJ) {
        // issue next K chunk into the other buffer, then wait for current:
        // outstanding = {chunk j, chunk j+1}; in-order => tensorcnt<=1 means
        // chunk j's DMA has landed in LDS.
        tdm_load_4kb(((j + 1) & 1) ? ks_lds1 : ks_lds0,
                     kp + (size_t)(j + 1) * 32 * D_);
        __builtin_amdgcn_s_wait_tensorcnt(1);
      } else {
        __builtin_amdgcn_s_wait_tensorcnt(0);
      }
    }
    __syncthreads();

    const _Float16* Ks = &Ks2[j & 1][0];

    // S = Q(16x64) * K^T(64x32): two 16-col tiles, K-dim split 2x32
    v8f s0 = {}, s1 = {};
    s0 = wmma32(Aq0, load_b_colmaj_f16(Ks, 64, 0, 0),  s0);
    s0 = wmma32(Aq1, load_b_colmaj_f16(Ks, 64, 0, 32), s0);
    s1 = wmma32(Aq0, load_b_colmaj_f16(Ks, 64, 16, 0),  s1);
    s1 = wmma32(Aq1, load_b_colmaj_f16(Ks, 64, 16, 32), s1);

    float corr[8];
#pragma unroll
    for (int i = 0; i < 8; ++i) {
      float mx = fmaxf(s0[i], s1[i]);
#pragma unroll
      for (int m = 1; m <= 8; m <<= 1) mx = fmaxf(mx, __shfl_xor(mx, m, 32));
      const float mnew = fmaxf(mrow[i], mx);
      corr[i] = __expf(mrow[i] - mnew);
      mrow[i] = mnew;
      s0[i] = __expf(s0[i] - mnew);
      s1[i] = __expf(s1[i] - mnew);
      float rs = s0[i] + s1[i];
#pragma unroll
      for (int m = 1; m <= 8; m <<= 1) rs += __shfl_xor(rs, m, 32);
      lrow[i] = lrow[i] * corr[i] + rs;
    }
#pragma unroll
    for (int t = 0; t < 4; ++t)
#pragma unroll
      for (int i = 0; i < 8; ++i) O[t][i] *= corr[i];

    // re-shape P (C/D layout -> A layout) through per-wave LDS
    _Float16* pw = &Ps[wave][0];
#pragma unroll
    for (int i = 0; i < 8; ++i) {
      pw[(i + lh * 8) * 32 + lr]      = (_Float16)s0[i];
      pw[(i + lh * 8) * 32 + 16 + lr] = (_Float16)s1[i];
    }
    const v16h Ap = load_a_f16(pw, 32, 0, 0);
#pragma unroll
    for (int t = 0; t < 4; ++t)
      O[t] = wmma32(Ap, load_b_colmaj_f16(Vt, 32, 16 * t, 0), O[t]);

    __syncthreads();   // all waves done with Vt / Ks2[j&1] before reuse
  }

  // normalize and emit in the reference's scrambled layout:
  // Z[b, d*32 + h*2 + (n>>10), n & 1023]
#pragma unroll
  for (int i = 0; i < 8; ++i) {
    const float inv = 1.0f / lrow[i];
    const int n = qbase + i + lh * 8;
#pragma unroll
    for (int t = 0; t < 4; ++t) {
      const int d  = 16 * t + lr;
      const int zr = d * 32 + h * 2 + (n >> 10);
      Z[((size_t)(b * N_ + zr)) * C_ + (n & (C_ - 1))] = (_Float16)(O[t][i] * inv);
    }
  }
}

// ---- K3: output projection + bias, fp32 out ---------------------------------
// grid (BN/128, C/64), block 256 (8 waves x 16 rows, 64 cols per wave)
__global__ void __launch_bounds__(256)
out_proj_kernel(const _Float16* __restrict__ Z, const float* __restrict__ Pw,
                const float* __restrict__ Pb, float* __restrict__ out) {
  const int wave = threadIdx.x >> 5;
  const int lane = threadIdx.x & 31;
  const int lh = lane >> 4, lr = lane & 15;
  const int rowbase = blockIdx.x * 128 + wave * 16;
  const int colbase = blockIdx.y * 64;

  v8f acc[4] = {};
  for (int k0 = 0; k0 < C_; k0 += 32) {
    v16h A = load_a_f16(Z, C_, rowbase, k0);
#pragma unroll
    for (int t = 0; t < 4; ++t)
      acc[t] = wmma32(A, load_b_colmaj_f32(Pw, C_, colbase + 16 * t, k0), acc[t]);
  }
#pragma unroll
  for (int t = 0; t < 4; ++t) {
    const float bias = Pb[colbase + 16 * t + lr];
#pragma unroll
    for (int i = 0; i < 8; ++i)
      out[(size_t)(rowbase + i + lh * 8) * C_ + colbase + 16 * t + lr] = acc[t][i] + bias;
  }
}

extern "C" void kernel_launch(void* const* d_in, const int* in_sizes, int n_in,
                              void* d_out, int out_size, void* d_ws, size_t ws_size,
                              hipStream_t stream) {
  const float* x      = (const float*)d_in[0];
  const float* y      = (const float*)d_in[1];
  const float* q_w    = (const float*)d_in[2];
  const float* kv_w   = (const float*)d_in[3];
  const float* proj_w = (const float*)d_in[4];
  const float* proj_b = (const float*)d_in[5];
  const float* temp   = (const float*)d_in[6];
  float* out = (float*)d_out;

  // workspace layout (all f16): Qn | Kn | Vh  (B,H,N,D each) | Z (B,N,C)
  const size_t elems = (size_t)B_ * H_ * N_ * D_;   // 8 Mi halves = 16 MB
  _Float16* Qn = (_Float16*)d_ws;
  _Float16* Kn = Qn + elems;
  _Float16* Vh = Kn + elems;
  _Float16* Z  = Vh + elems;                        // also 8 Mi halves

  proj_qkv_kernel<<<dim3(BN_ / 64, H_, 3), 128, 0, stream>>>(x, y, q_w, kv_w, temp,
                                                             Qn, Kn, Vh);
  xca_attn_kernel<<<dim3(N_ / 128, H_, B_), 256, 0, stream>>>(Qn, Kn, Vh, Z);
  out_proj_kernel<<<dim3(BN_ / 128, C_ / 64), 256, 0, stream>>>(Z, proj_w, proj_b, out);
}
